// SimpleRNN_40724879900895
// MI455X (gfx1250) — matrix operand order repaired and checked by reading, not compiled, not run on hardware
//
#include <hip/hip_runtime.h>

#define Hdim 2048
#define Bdim 1024
#define Vdim 8192
#define Ldim 8

#define BM 128
#define BN 128
#define BK 32
#define PAD 8       // halves of padding per LDS row (row stride 80B)
#define NSTAGE 4    // async pipeline depth; uniform wait: asynccnt <= 8
#define KSTEPS (Hdim / BK)   // 64

typedef __attribute__((ext_vector_type(16))) _Float16 v16h;
typedef __attribute__((ext_vector_type(8)))  _Float16 v8h;
typedef __attribute__((ext_vector_type(8)))  float    v8f;

// ---- CDNA5 async global->LDS via inline asm -------------------------------
__device__ __forceinline__ unsigned lds_addr32(const void* p) {
    // flat LDS aperture: LDS byte address is the low 32 bits of the flat addr
    return (unsigned)(unsigned long long)p;
}

__device__ __forceinline__ void async_load_b128(const void* gsrc, void* ldst) {
    asm volatile("global_load_async_to_lds_b128 %0, %1, off"
                 :: "v"(lds_addr32(ldst)), "v"(gsrc) : "memory");
}

#if __has_builtin(__builtin_amdgcn_s_wait_asynccnt)
#define WAIT_ASYNC(n) __builtin_amdgcn_s_wait_asynccnt(n)
#else
#define WAIT_ASYNC(n) asm volatile("s_wait_asynccnt " #n ::: "memory")
#endif

#if __has_builtin(__builtin_amdgcn_tanhf)
#define TANH(x) __builtin_amdgcn_tanhf(x)
#elif __has_builtin(__builtin_amdgcn_tanh_f32)
#define TANH(x) __builtin_amdgcn_tanh_f32(x)
#else
#define TANH(x) tanhf(x)
#endif

// ---------------------------------------------------------------------------
// fp32 -> f16 state conversion
// ---------------------------------------------------------------------------
__global__ void cvt_state_kernel(const float* __restrict__ src,
                                 _Float16* __restrict__ dst) {
    int i = blockIdx.x * blockDim.x + threadIdx.x;
    dst[i] = (_Float16)src[i];
}

// ---------------------------------------------------------------------------
// hh[l][k][n] fp32 -> hhT[l][n][k] f16  (tiled transpose through LDS)
// grid: (H/32, H/32, L), block: (32, 8)
// ---------------------------------------------------------------------------
__global__ void cvt_hh_kernel(const float* __restrict__ hh,
                              _Float16* __restrict__ hhT) {
    __shared__ float tile[32][33];
    const int l  = blockIdx.z;
    const int n0 = blockIdx.x * 32;
    const int k0 = blockIdx.y * 32;
    const float*    src = hh  + (size_t)l * Hdim * Hdim;
    _Float16*       dst = hhT + (size_t)l * Hdim * Hdim;
    const int tx = threadIdx.x;   // 0..31
    const int ty = threadIdx.y;   // 0..7
#pragma unroll
    for (int i = 0; i < 32; i += 8)
        tile[ty + i][tx] = src[(size_t)(k0 + ty + i) * Hdim + (n0 + tx)];
    __syncthreads();
#pragma unroll
    for (int i = 0; i < 32; i += 8)
        dst[(size_t)(n0 + ty + i) * Hdim + (k0 + tx)] = (_Float16)tile[tx][ty + i];
}

// ---------------------------------------------------------------------------
// One RNN layer:  sNew = tanh(sOld + (s @ hh) * ih[token])
// grid: (H/BN, B/BM), block: 256 (8 waves of 32), 4-stage async LDS pipeline
// ---------------------------------------------------------------------------
__global__ __launch_bounds__(256)
void rnn_layer_kernel(const float*    __restrict__ sOld,
                      const _Float16* __restrict__ sHalf,
                      const _Float16* __restrict__ hhT,
                      const float*    __restrict__ ihL,
                      const long long* __restrict__ token,
                      float*    __restrict__ sNew,
                      _Float16* __restrict__ sNewHalf) {
    __shared__ _Float16 As[NSTAGE][BM][BK + PAD];
    __shared__ _Float16 Bs[NSTAGE][BN][BK + PAD];

    const int tid  = threadIdx.x;
    const int wave = tid >> 5;       // 0..7
    const int lane = tid & 31;
    const int wm   = wave >> 2;      // 0..1 : rows  wm*64 .. +63
    const int wn   = wave & 3;       // 0..3 : cols  wn*32 .. +31
    const int hs   = lane >> 4;      // half-wave select
    const int l16  = lane & 15;

    const int mBlock = blockIdx.y * BM;
    const int nBlock = blockIdx.x * BN;

    v8f acc[4][2] = {};

    // Stage one BMxBK A-tile and BNxBK B-tile into LDS stage `buf`.
    // 4 async-to-LDS instructions per thread per stage (ASYNCcnt += 4).
    auto stage = [&](int buf, int kidx) {
        const int k0 = kidx * BK;
#pragma unroll
        for (int i = 0; i < 2; i++) {
            const int c  = tid + i * 256;
            const int r  = c >> 2;          // 0..127
            const int kk = (c & 3) * 8;     // 0,8,16,24
            async_load_b128(&sHalf[(size_t)(mBlock + r) * Hdim + k0 + kk],
                            &As[buf][r][kk]);
            async_load_b128(&hhT[(size_t)(nBlock + r) * Hdim + k0 + kk],
                            &Bs[buf][r][kk]);
        }
    };

    // One K-step of WMMAs from LDS stage `curb`.
    auto mma_step = [&](int curb) {
        v16h bF[2];
#pragma unroll
        for (int ni = 0; ni < 2; ni++) {
            const int col = wn * 32 + ni * 16 + l16;
            union { v16h v; v8h h[2]; } u;
            u.h[0] = *(const v8h*)&Bs[curb][col][hs * 16];
            u.h[1] = *(const v8h*)&Bs[curb][col][hs * 16 + 8];
            bF[ni] = u.v;
        }
        v16h aF[4];
#pragma unroll
        for (int mi = 0; mi < 4; mi++) {
            const int row = wm * 64 + mi * 16 + l16;
            union { v16h v; v8h h[2]; } u;
            u.h[0] = *(const v8h*)&As[curb][row][hs * 16];
            u.h[1] = *(const v8h*)&As[curb][row][hs * 16 + 8];
            aF[mi] = u.v;
        }
#pragma unroll
        for (int mi = 0; mi < 4; mi++)
#pragma unroll
            for (int ni = 0; ni < 2; ni++)
                acc[mi][ni] = __builtin_amdgcn_wmma_f32_16x16x32_f16(
                    false, aF[mi], false, bF[ni], (short)0, acc[mi][ni],
                    false, false);
    };

    // Prologue: fill NSTAGE-1 pipeline stages.
    stage(0, 0);
    stage(1, 1);
    stage(2, 2);

    // Uniform steady-state loop: always 3 stages in flight (asynccnt==12 at
    // the wait), so WAIT_ASYNC(8) retires exactly the stage consumed now.
    // Final 3 iterations prefetch the last K-tile again (L2-hot, never read)
    // purely to keep the counter arithmetic -- and thus the codegen -- uniform.
#pragma unroll 4
    for (int idx = 0; idx < KSTEPS; idx++) {
        WAIT_ASYNC(8);
        __syncthreads();    // stage idx visible to all; reads of idx-1 retired
        int kidx = idx + 3;
        if (kidx > KSTEPS - 1) kidx = KSTEPS - 1;   // dummy tail prefetch
        stage((idx + 3) & 3, kidx);
        mma_step(idx & 3);
    }
    // No drain needed: epilogue touches no LDS; S_ENDPGM implies wait-idle.

    // Epilogue: C/D layout -> VGPR r is row (r + 8*hs), col = l16.
    // ih-gather and sOld are single-use streaming reads -> nontemporal.
#pragma unroll
    for (int mi = 0; mi < 4; mi++) {
#pragma unroll
        for (int r = 0; r < 8; r++) {
            const int row = mBlock + wm * 64 + mi * 16 + hs * 8 + r;
            const long long tok = token[row];
            const float* gateRow = ihL + (size_t)tok * Hdim;
            const float* oldRow  = sOld + (size_t)row * Hdim;
            float*       newRow  = sNew + (size_t)row * Hdim;
            _Float16*    newHRow = sNewHalf + (size_t)row * Hdim;
#pragma unroll
            for (int ni = 0; ni < 2; ni++) {
                const int col = nBlock + wn * 32 + ni * 16 + l16;
                const float g = __builtin_nontemporal_load(&gateRow[col]);
                const float o = __builtin_nontemporal_load(&oldRow[col]);
                const float t = TANH(o + acc[mi][ni][r] * g);
                newRow[col]  = t;
                newHRow[col] = (_Float16)t;
            }
        }
    }
}

// ---------------------------------------------------------------------------
extern "C" void kernel_launch(void* const* d_in, const int* in_sizes, int n_in,
                              void* d_out, int out_size, void* d_ws, size_t ws_size,
                              hipStream_t stream) {
    const float*     state = (const float*)d_in[0];
    const long long* token = (const long long*)d_in[1];
    const float*     ih    = (const float*)d_in[2];
    const float*     hh    = (const float*)d_in[3];

    char* ws = (char*)d_ws;
    const size_t hhT_bytes = (size_t)Ldim * Hdim * Hdim * sizeof(_Float16); // 64 MB
    const size_t s16_bytes = (size_t)Bdim * Hdim * sizeof(_Float16);        // 4 MB
    const size_t s32_bytes = (size_t)Bdim * Hdim * sizeof(float);           // 8 MB

    _Float16* hhT  = (_Float16*)ws;
    _Float16* s16a = (_Float16*)(ws + hhT_bytes);
    _Float16* s16b = (_Float16*)(ws + hhT_bytes + s16_bytes);
    float*    f32a = (float*)(ws + hhT_bytes + 2 * s16_bytes);
    float*    f32b = (float*)(ws + hhT_bytes + 2 * s16_bytes + s32_bytes);

    cvt_state_kernel<<<(Bdim * Hdim) / 256, 256, 0, stream>>>(state, s16a);
    cvt_hh_kernel<<<dim3(Hdim / 32, Hdim / 32, Ldim), dim3(32, 8), 0, stream>>>(hh, hhT);

    const float*    inF = state;
    const _Float16* inH = s16a;
    dim3 grid(Hdim / BN, Bdim / BM);
    for (int l = 0; l < Ldim; l++) {
        float*    outF = (l == Ldim - 1) ? (float*)d_out : ((l & 1) ? f32b : f32a);
        _Float16* outH = (l & 1) ? s16a : s16b;
        rnn_layer_kernel<<<grid, 256, 0, stream>>>(
            inF, inH,
            hhT + (size_t)l * Hdim * Hdim,
            ih  + (size_t)l * Vdim * Hdim,
            token, outF, outH);
        inF = outF;
        inH = outH;
    }
}
